// GraphConsis_17910013624527
// MI455X (gfx1250) — matrix operand order repaired and checked by reading, not compiled, not run on hardware
//
#include <hip/hip_runtime.h>

typedef _Float16 v16h __attribute__((ext_vector_type(16)));
typedef _Float16 v8h  __attribute__((ext_vector_type(8)));
typedef _Float16 v4h  __attribute__((ext_vector_type(4)));
typedef float    v8f  __attribute__((ext_vector_type(8)));

#define NMAT 8192
#define DDIM 128
#define BT   128   // block tile (M and N)
#define LDT  136   // padded LDS row stride in halves (272 B)

// ---------------------------------------------------------------- init
__global__ void init_accum_kernel(void* ws) {
    double* d = (double*)ws;
    d[0] = 0.0;   // sum(dist * gw)
    d[1] = 0.0;   // sum(gw)
    *(unsigned int*)((char*)ws + 16) = 0u;  // max(dist) as uint bits
}

// ---------------------------------------------------------------- sq norms
__global__ __launch_bounds__(128) void sqnorm_kernel(const float* __restrict__ emb,
                                                     float* __restrict__ sqn) {
    __shared__ float part[4];
    const int row = blockIdx.x;
    const int t   = threadIdx.x;
    float v = emb[(size_t)row * DDIM + t];
    v *= v;
    for (int o = 16; o > 0; o >>= 1) v += __shfl_down(v, o, 32);
    if ((t & 31) == 0) part[t >> 5] = v;
    __syncthreads();
    if (t == 0) sqn[row] = part[0] + part[1] + part[2] + part[3];
}

// ---------------------------------------------------------------- helpers
__device__ __forceinline__ v8f wmma16(v16h a, v16h b, v8f c) {
    return __builtin_amdgcn_wmma_f32_16x16x32_f16(false, a, false, b, (short)0, c,
                                                  false, false);
}

// A-frag (16x32 f16): lane L<16 holds row M=L, halves[0..7]=K k0..k0+7,
// halves[8..15]=K k0+16..k0+23 with k0 = kc + 8*(L>=16).
__device__ __forceinline__ v16h load_a_frag(const _Float16* p) {
    v8h a0 = *(const v8h*)p;
    v8h a1 = *(const v8h*)(p + 16);
    return __builtin_shufflevector(a0, a1, 0, 1, 2, 3, 4, 5, 6, 7,
                                           8, 9, 10, 11, 12, 13, 14, 15);
}

// B-frag (32x16 f16): lane L holds column N=L&15, halves[0..7]=K k0..k0+7,
// halves[8..15]=K k0+16..k0+23 with k0 = kc + 8*(L>=16).
__device__ __forceinline__ v16h load_b_frag(const _Float16* p) {
    v8h b0 = *(const v8h*)p;
    v8h b1 = *(const v8h*)(p + 16);
    return __builtin_shufflevector(b0, b1, 0, 1, 2, 3, 4, 5, 6, 7,
                                           8, 9, 10, 11, 12, 13, 14, 15);
}

// fp32 -> (hi, lo) f16 split of a float4. hi = rn(x); lo = rn(x - hi).
__device__ __forceinline__ void split4(float4 v, v4h& h, v4h& l) {
    const _Float16 h0 = (_Float16)v.x;
    const _Float16 h1 = (_Float16)v.y;
    const _Float16 h2 = (_Float16)v.z;
    const _Float16 h3 = (_Float16)v.w;
    h = (v4h){h0, h1, h2, h3};
    l = (v4h){(_Float16)(v.x - (float)h0), (_Float16)(v.y - (float)h1),
              (_Float16)(v.z - (float)h2), (_Float16)(v.w - (float)h3)};
}

// ---------------------------------------------------------------- fused GEMM+loss
__global__ __launch_bounds__(256) void graph_loss_kernel(
    const float* __restrict__ emb,
    const float* __restrict__ gw,
    const float* __restrict__ sqn,
    double* __restrict__ sums,
    unsigned int* __restrict__ maxbits) {

    extern __shared__ _Float16 sm[];
    _Float16* Ahi = sm;
    _Float16* Alo = Ahi + BT * LDT;
    _Float16* Bhi = Alo + BT * LDT;
    _Float16* Blo = Bhi + BT * LDT;

    const int tid     = threadIdx.x;
    const int rowBase = blockIdx.x * BT;
    const int colBase = blockIdx.y * BT;

    // Prefetch the graphWeight tile (streams from HBM while we do the GEMM).
    // 128 rows x 512 B = 4 x 128B lines per row -> 512 lines.
    for (int i = tid; i < 512; i += 256) {
        const float* p = &gw[(size_t)(rowBase + (i >> 2)) * NMAT + colBase + (i & 3) * 32];
        __builtin_prefetch(p, 0, 0);
    }

    // Cooperative load of both emb tiles with fp32 -> (hi,lo) f16 split.
    for (int i = tid; i < BT * 32; i += 256) {
        const int r  = i >> 5;
        const int c4 = (i & 31) << 2;
        float4 va = *(const float4*)&emb[(size_t)(rowBase + r) * DDIM + c4];
        float4 vb = *(const float4*)&emb[(size_t)(colBase + r) * DDIM + c4];
        v4h h, l;
        split4(va, h, l);
        *(v4h*)&Ahi[r * LDT + c4] = h;
        *(v4h*)&Alo[r * LDT + c4] = l;
        split4(vb, h, l);
        *(v4h*)&Bhi[r * LDT + c4] = h;
        *(v4h*)&Blo[r * LDT + c4] = l;
    }
    __syncthreads();

    const int lane   = tid & 31;
    const int wid    = tid >> 5;
    const int wm     = wid >> 2;   // 0..1 : 64 rows each
    const int wn     = wid & 3;    // 0..3 : 32 cols each
    const int laneLo = lane & 15;
    const int laneHi = lane >> 4;

    v8f c[4][2];
#pragma unroll
    for (int mt = 0; mt < 4; ++mt)
#pragma unroll
        for (int nt = 0; nt < 2; ++nt) c[mt][nt] = (v8f)(0.0f);

#pragma unroll
    for (int kc = 0; kc < DDIM; kc += 32) {
        v16h afh[4], afl[4], bfh[2], bfl[2];
        const int ka = kc + 8 * laneHi;
        const int kb = kc + 8 * laneHi;
#pragma unroll
        for (int mt = 0; mt < 4; ++mt) {
            const int row = wm * 64 + mt * 16 + laneLo;
            afh[mt] = load_a_frag(&Ahi[row * LDT + ka]);
            afl[mt] = load_a_frag(&Alo[row * LDT + ka]);
        }
#pragma unroll
        for (int nt = 0; nt < 2; ++nt) {
            const int crow = wn * 32 + nt * 16 + laneLo;
            bfh[nt] = load_b_frag(&Bhi[crow * LDT + kb]);
            bfl[nt] = load_b_frag(&Blo[crow * LDT + kb]);
        }
#pragma unroll
        for (int mt = 0; mt < 4; ++mt)
#pragma unroll
            for (int nt = 0; nt < 2; ++nt) {
                c[mt][nt] = wmma16(afh[mt], bfh[nt], c[mt][nt]);  // hi*hi
                c[mt][nt] = wmma16(afh[mt], bfl[nt], c[mt][nt]);  // hi*lo
                c[mt][nt] = wmma16(afl[mt], bfh[nt], c[mt][nt]);  // lo*hi
            }
    }

    // Epilogue: dist + fused weighted reduction. C layout: value i of a tile
    // lives at row = i + 8*laneHi, col = laneLo (ISA 7.12.2).
    float accDW = 0.0f, accW = 0.0f, mx = 0.0f;
#pragma unroll
    for (int mt = 0; mt < 4; ++mt) {
        const int rbase = rowBase + wm * 64 + mt * 16 + 8 * laneHi;
        float rn[8];
#pragma unroll
        for (int i = 0; i < 8; ++i) rn[i] = sqn[rbase + i];
#pragma unroll
        for (int nt = 0; nt < 2; ++nt) {
            const int col  = colBase + wn * 32 + nt * 16 + laneLo;
            const float cn = sqn[col];
            const float* gp = &gw[(size_t)rbase * NMAT + col];
#pragma unroll
            for (int i = 0; i < 8; ++i) {
                const float dot = c[mt][nt][i];
                float sq = rn[i] + cn - 2.0f * dot;
                sq = fmaxf(sq, 0.0f);
                const float d = sqrtf(sq);
                const float g = gp[(size_t)i * NMAT];
                accDW = fmaf(d, g, accDW);
                accW += g;
                mx = fmaxf(mx, d);
            }
        }
    }

    // Wave reduction.
    for (int o = 16; o > 0; o >>= 1) {
        accDW += __shfl_down(accDW, o, 32);
        accW  += __shfl_down(accW, o, 32);
        mx = fmaxf(mx, __shfl_down(mx, o, 32));
    }

    __syncthreads();                 // all LDS fragment reads finished
    float* red = (float*)sm;         // reuse LDS for cross-wave reduction
    if (lane == 0) {
        red[wid]      = accDW;
        red[8 + wid]  = accW;
        red[16 + wid] = mx;
    }
    __syncthreads();
    if (tid == 0) {
        float s_dw = 0.0f, s_w = 0.0f, m = 0.0f;
        for (int w = 0; w < 8; ++w) {
            s_dw += red[w];
            s_w  += red[8 + w];
            m = fmaxf(m, red[16 + w]);
        }
        atomicAdd(&sums[0], (double)s_dw);
        atomicAdd(&sums[1], (double)s_w);
        atomicMax(maxbits, __float_as_uint(m));  // valid: dist >= 0
    }
}

// ---------------------------------------------------------------- finalize
__global__ void finalize_kernel(const double* __restrict__ sums,
                                const unsigned int* __restrict__ maxbits,
                                float* __restrict__ out) {
    const double mx = (double)__uint_as_float(*maxbits);
    const double r  = (sums[1] - sums[0] / mx) / ((double)NMAT * (double)NMAT);
    out[0] = (float)r;
}

// ---------------------------------------------------------------- launch
extern "C" void kernel_launch(void* const* d_in, const int* in_sizes, int n_in,
                              void* d_out, int out_size, void* d_ws, size_t ws_size,
                              hipStream_t stream) {
    (void)in_sizes; (void)n_in; (void)out_size; (void)ws_size;
    const float* emb = (const float*)d_in[0];
    const float* gw  = (const float*)d_in[1];
    float* out = (float*)d_out;

    double*       sums    = (double*)d_ws;
    unsigned int* maxbits = (unsigned int*)((char*)d_ws + 16);
    float*        sqn     = (float*)((char*)d_ws + 32);

    hipLaunchKernelGGL(init_accum_kernel, dim3(1), dim3(1), 0, stream, d_ws);
    hipLaunchKernelGGL(sqnorm_kernel, dim3(NMAT), dim3(128), 0, stream, emb, sqn);

    const size_t shmem = (size_t)4 * BT * LDT * sizeof(_Float16);  // 139264 B
    hipLaunchKernelGGL(graph_loss_kernel, dim3(NMAT / BT, NMAT / BT), dim3(256),
                       shmem, stream, emb, gw, sqn, sums, maxbits);

    hipLaunchKernelGGL(finalize_kernel, dim3(1), dim3(1), 0, stream, sums, maxbits, out);
}
